// RGCNLayer_59742995087718
// MI455X (gfx1250) — matrix operand order repaired and checked by reading, not compiled, not run on hardware
//
#include <hip/hip_runtime.h>
#include <math.h>

#define HIDDEN 512
#define NNODE  256
#define BSZ    32
#define RELS   16
#define KTOT   (RELS * HIDDEN)   // 8192
#define KSTEPS (KTOT / 32)       // 256

typedef __attribute__((ext_vector_type(16))) __bf16 v16bf;
typedef __attribute__((ext_vector_type(8)))  float  v8f;

union BF16x16 { v16bf v; unsigned short u[16]; };
union U4x8    { uint4 q; unsigned short s[8]; };

static __device__ __forceinline__ unsigned short f2bf(float f) {
    union { float f; unsigned int u; } x;
    x.f = f;
    unsigned int r = x.u + 0x7FFFu + ((x.u >> 16) & 1u); // round-to-nearest-even
    return (unsigned short)(r >> 16);
}

// ---------------------------------------------------------------------------
// Kernel 1: h = concat(columns, logits) @ W_proj + b ; LayerNorm ; exact GELU
// One block (256 threads) per row (b*256+n). Emits x fp32 + x bf16.
// ---------------------------------------------------------------------------
__global__ __launch_bounds__(256) void proj_ln_gelu_kernel(
    const float* __restrict__ columns,   // [B][N][512]
    const float* __restrict__ logits,    // [B][N][2]
    const float* __restrict__ W_proj,    // [514][512]
    const float* __restrict__ b_proj,    // [512]
    const float* __restrict__ ln_g,
    const float* __restrict__ ln_b,
    float* __restrict__ x_f32,           // [B][N][512]
    unsigned short* __restrict__ x_bf)   // [B][N][512]
{
    __shared__ float inrow[514];
    __shared__ float red[256];
    const int row = blockIdx.x;
    const int tid = threadIdx.x;

    const float* crow = &columns[(size_t)row * HIDDEN];
    inrow[tid]        = crow[tid];
    inrow[tid + 256]  = crow[tid + 256];
    if (tid < 2) inrow[512 + tid] = logits[(size_t)row * 2 + tid];
    __syncthreads();

    float a0 = b_proj[tid];
    float a1 = b_proj[tid + 256];
    for (int i = 0; i < 514; ++i) {
        const float xi = inrow[i];
        const float* wp = &W_proj[(size_t)i * HIDDEN];
        a0 = fmaf(xi, wp[tid], a0);
        a1 = fmaf(xi, wp[tid + 256], a1);
    }

    red[tid] = a0 + a1;
    __syncthreads();
    for (int s = 128; s > 0; s >>= 1) {
        if (tid < s) red[tid] += red[tid + s];
        __syncthreads();
    }
    const float mu = red[0] * (1.0f / 512.0f);
    __syncthreads();

    const float d0 = a0 - mu, d1 = a1 - mu;
    red[tid] = d0 * d0 + d1 * d1;
    __syncthreads();
    for (int s = 128; s > 0; s >>= 1) {
        if (tid < s) red[tid] += red[tid + s];
        __syncthreads();
    }
    const float rstd = rsqrtf(red[0] * (1.0f / 512.0f) + 1e-5f);

    const float h0 = d0 * rstd * ln_g[tid]       + ln_b[tid];
    const float h1 = d1 * rstd * ln_g[tid + 256] + ln_b[tid + 256];
    const float g0 = 0.5f * h0 * (1.0f + erff(h0 * 0.70710678118654752f));
    const float g1 = 0.5f * h1 * (1.0f + erff(h1 * 0.70710678118654752f));

    const size_t base = (size_t)row * HIDDEN;
    x_f32[base + tid]       = g0;
    x_f32[base + tid + 256] = g1;
    x_bf[base + tid]        = f2bf(g0);
    x_bf[base + tid + 256]  = f2bf(g1);
}

// ---------------------------------------------------------------------------
// Kernel 2: Wr[r,h,o] = sum_t V[r,t]*W[h,t,o], emitted PRE-TRANSPOSED:
//   wr_t[o][k] with k = r*512+h   -> bf16 [512][8192]
// Block handles a 32(k) x 64(o) tile; transpose through LDS so global writes
// are 16B-contiguous along k.
// ---------------------------------------------------------------------------
__global__ __launch_bounds__(256) void mix_wr_t_kernel(
    const float* __restrict__ W,           // [512][3][512]
    const float* __restrict__ V,           // [16][3]
    unsigned short* __restrict__ wr_t)     // [512][8192]
{
    __shared__ unsigned short lT[32][72];  // [k_local][o_local], padded
    const int rh0 = blockIdx.x * 32;       // k tile base (always within one r)
    const int o0  = blockIdx.y * 64;
    const int tid = threadIdx.x;

    const int r  = rh0 >> 9;
    const int h0 = rh0 & 511;
    const float v0 = V[r * 3 + 0], v1 = V[r * 3 + 1], v2 = V[r * 3 + 2];

    // compute phase: thread -> (h_local = tid>>3, 8 contiguous o)
    {
        const int hl = tid >> 3;
        const int oc = (tid & 7) * 8;
        const float* wb = &W[((size_t)(h0 + hl) * 3) * HIDDEN + o0 + oc];
#pragma unroll
        for (int e = 0; e < 8; ++e) {
            const float s = v0 * wb[e] + v1 * wb[HIDDEN + e] + v2 * wb[2 * HIDDEN + e];
            lT[hl][oc + e] = f2bf(s);
        }
    }
    __syncthreads();

    // write phase: thread -> (o_local = tid>>2, 8 contiguous k)
    {
        const int ol = tid >> 2;
        const int kc = (tid & 3) * 8;
        U4x8 t;
#pragma unroll
        for (int e = 0; e < 8; ++e) t.s[e] = lT[kc + e][ol];
        *(uint4*)&wr_t[(size_t)(o0 + ol) * KTOT + rh0 + kc] = t.q;
    }
}

// ---------------------------------------------------------------------------
// Kernel 3: sup[g, r-1, n, :] = sum_m [adj[b,n,m]==r] * x[b,m,:]  (r=1..15)
// ---------------------------------------------------------------------------
__global__ __launch_bounds__(256) void aggregate_kernel(
    const float* __restrict__ x,            // [B][N][512]
    const int*   __restrict__ adj,          // [B][N][N]
    unsigned short* __restrict__ sup_bf,    // [G][15][N][512]
    int b_base)
{
    __shared__ float s[15 * HIDDEN];        // 30720 B
    const int n = blockIdx.x;
    const int g = blockIdx.y;
    const int b = b_base + g;
    const int tid = threadIdx.x;

    for (int i = tid; i < 15 * HIDDEN; i += 256) s[i] = 0.0f;
    __syncthreads();

    const int* arow = &adj[((size_t)b * NNODE + n) * NNODE];
    for (int m = 0; m < NNODE; ++m) {
        const int r = arow[m];
        if (r > 0) {
            const float* xr = &x[((size_t)b * NNODE + m) * HIDDEN];
            const int base = (r - 1) * HIDDEN;
            s[base + tid]       += xr[tid];
            s[base + tid + 256] += xr[tid + 256];
        }
    }
    __syncthreads();

    for (int i = tid; i < 15 * HIDDEN; i += 256) {
        const int r = i >> 9;
        const int h = i & 511;
        sup_bf[((((size_t)g * 15 + r) * NNODE) + n) * HIDDEN + h] = f2bf(s[i]);
    }
}

// ---------------------------------------------------------------------------
// Kernel 4: out[b] = A_cat(256 x 8192) @ Wr_cat(8192 x 512), bf16 WMMA.
// Block tile 128(n) x 64(o); 8 waves in a 4x2 grid; each wave owns a 2x2 set
// of 16x16 WMMA tiles. Double-buffered LDS, one barrier per K-step.
// ---------------------------------------------------------------------------
__global__ __launch_bounds__(256) void rgcn_gemm_kernel(
    const unsigned short* __restrict__ x_bf,    // [B][N][512]
    const unsigned short* __restrict__ sup_bf,  // [G][15][N][512]
    const unsigned short* __restrict__ wr_t,    // [512][8192]
    float* __restrict__ out,                    // [B][N][512]
    int b_base)
{
    __shared__ alignas(16) unsigned short lA[2][128 * 40]; // A[row][k]
    __shared__ alignas(16) unsigned short lB[2][64 * 40];  // Bt[col][k]

    const int g  = blockIdx.y;
    const int b  = b_base + g;
    const int tx = blockIdx.x;            // 0..15
    const int n0 = (tx & 1) * 128;
    const int o0 = (tx >> 1) * 64;
    const int tid  = threadIdx.x;
    const int lane = tid & 31;
    const int wave = tid >> 5;
    const int wr_  = wave >> 1;           // wave row 0..3 (32 n each)
    const int wc_  = wave & 1;            // wave col 0..1 (32 o each)

    // global staging coordinates (fixed per thread)
    const int arow_s  = tid >> 1;         // 0..127
    const int ahb_s   = (tid & 1) * 16;   // 0 or 16
    const int bcol_s  = tid >> 2;         // 0..63
    const int bkoff_s = (tid & 3) * 8;    // 0,8,16,24

    // fragment coordinates
    const int kbase = (lane < 16) ? 0 : 8;
    const int koffB = (lane < 16) ? 0 : 16;
    const int l15   = lane & 15;
    const int ar0 = (wr_ * 32 + l15) * 40;
    const int ar1 = (wr_ * 32 + 16 + l15) * 40;
    const int bc0 = (wc_ * 32 + l15) * 40;
    const int bc1 = (wc_ * 32 + 16 + l15) * 40;

    v8f acc00 = {}, acc01 = {}, acc10 = {}, acc11 = {};

    auto srcA = [&](int kt) -> const unsigned short* {
        const int k0 = kt * 32;
        const int r  = k0 >> 9;
        const int h0 = k0 & 511;
        return (r == 0)
            ? &x_bf[(((size_t)b * NNODE) + (n0 + arow_s)) * HIDDEN + h0 + ahb_s]
            : &sup_bf[((((size_t)g * 15 + (r - 1)) * NNODE) + (n0 + arow_s)) * HIDDEN + h0 + ahb_s];
    };

    // prologue: stage K-step 0 into buffer 0
    {
        const unsigned short* sa = srcA(0);
        *(uint4*)&lA[0][arow_s * 40 + ahb_s]     = *(const uint4*)sa;
        *(uint4*)&lA[0][arow_s * 40 + ahb_s + 8] = *(const uint4*)(sa + 8);
        *(uint4*)&lB[0][bcol_s * 40 + bkoff_s] =
            *(const uint4*)&wr_t[(size_t)(o0 + bcol_s) * KTOT + bkoff_s];
    }
    __syncthreads();

    auto compute = [&](const unsigned short* A, const unsigned short* Bt) {
        BF16x16 a0, a1, b0, b1;
#pragma unroll
        for (int e = 0; e < 8; ++e) {
            a0.u[e]     = A[ar0 + kbase + e];
            a0.u[8 + e] = A[ar0 + kbase + 16 + e];
            a1.u[e]     = A[ar1 + kbase + e];
            a1.u[8 + e] = A[ar1 + kbase + 16 + e];
        }
#pragma unroll
        for (int e = 0; e < 8; ++e) {
            b0.u[e]     = Bt[bc0 + kbase + e];
            b0.u[8 + e] = Bt[bc0 + kbase + 16 + e];
            b1.u[e]     = Bt[bc1 + kbase + e];
            b1.u[8 + e] = Bt[bc1 + kbase + 16 + e];
        }
        acc00 = __builtin_amdgcn_wmma_f32_16x16x32_bf16(false, a0.v, false, b0.v,
                                                        (short)0, acc00, false, false);
        acc01 = __builtin_amdgcn_wmma_f32_16x16x32_bf16(false, a0.v, false, b1.v,
                                                        (short)0, acc01, false, false);
        acc10 = __builtin_amdgcn_wmma_f32_16x16x32_bf16(false, a1.v, false, b0.v,
                                                        (short)0, acc10, false, false);
        acc11 = __builtin_amdgcn_wmma_f32_16x16x32_bf16(false, a1.v, false, b1.v,
                                                        (short)0, acc11, false, false);
    };

    int cur = 0;
    for (int kt = 0; kt < KSTEPS - 1; ++kt) {
        // prefetch next K-step into registers
        const unsigned short* sa = srcA(kt + 1);
        const uint4 pa0 = *(const uint4*)sa;
        const uint4 pa1 = *(const uint4*)(sa + 8);
        const uint4 pb  = *(const uint4*)&wr_t[(size_t)(o0 + bcol_s) * KTOT
                                               + (kt + 1) * 32 + bkoff_s];
        // compute current buffer
        compute(lA[cur], lB[cur]);
        // store prefetch into other buffer (no reader until after barrier)
        const int nxt = cur ^ 1;
        *(uint4*)&lA[nxt][arow_s * 40 + ahb_s]     = pa0;
        *(uint4*)&lA[nxt][arow_s * 40 + ahb_s + 8] = pa1;
        *(uint4*)&lB[nxt][bcol_s * 40 + bkoff_s]   = pb;
        __syncthreads();
        cur = nxt;
    }
    compute(lA[cur], lB[cur]);

    // epilogue: C/D layout -> VGPR v: lanes<16 row +v, lanes>=16 row +8+v
    const int rlo = ((lane >> 4) << 3);
#pragma unroll
    for (int v = 0; v < 8; ++v) {
        const int row0 = n0 + wr_ * 32 + rlo + v;
        const size_t base0 = (((size_t)b * NNODE) + row0) * HIDDEN;
        const size_t base1 = base0 + (size_t)16 * HIDDEN;
        out[base0 + o0 + wc_ * 32 + l15]      = acc00[v];
        out[base0 + o0 + wc_ * 32 + 16 + l15] = acc01[v];
        out[base1 + o0 + wc_ * 32 + l15]      = acc10[v];
        out[base1 + o0 + wc_ * 32 + 16 + l15] = acc11[v];
    }
}

// ---------------------------------------------------------------------------
extern "C" void kernel_launch(void* const* d_in, const int* in_sizes, int n_in,
                              void* d_out, int out_size, void* d_ws, size_t ws_size,
                              hipStream_t stream) {
    const float* columns = (const float*)d_in[0];
    const float* logits  = (const float*)d_in[1];
    const int*   adj     = (const int*)  d_in[2];
    const float* W_proj  = (const float*)d_in[3];
    const float* b_proj  = (const float*)d_in[4];
    const float* ln_g    = (const float*)d_in[5];
    const float* ln_b    = (const float*)d_in[6];
    const float* W       = (const float*)d_in[7];
    const float* V       = (const float*)d_in[8];
    float* out = (float*)d_out;

    char* ws = (char*)d_ws;
    size_t off = 0;
    float* x_f32 = (float*)(ws + off);
    off += (size_t)BSZ * NNODE * HIDDEN * sizeof(float);           // 16.78 MB
    unsigned short* x_bf = (unsigned short*)(ws + off);
    off += (size_t)BSZ * NNODE * HIDDEN * sizeof(unsigned short);  //  8.39 MB
    unsigned short* wr_t = (unsigned short*)(ws + off);
    off += (size_t)HIDDEN * KTOT * sizeof(unsigned short);         //  8.39 MB
    unsigned short* sup_bf = (unsigned short*)(ws + off);

    const size_t per_b = (size_t)15 * NNODE * HIDDEN * sizeof(unsigned short); // 3.93 MB
    size_t remain = (ws_size > off) ? (ws_size - off) : per_b;
    int G = (int)(remain / per_b);
    if (G < 1)   G = 1;
    if (G > BSZ) G = BSZ;

    proj_ln_gelu_kernel<<<BSZ * NNODE, 256, 0, stream>>>(
        columns, logits, W_proj, b_proj, ln_g, ln_b, x_f32, x_bf);
    mix_wr_t_kernel<<<dim3(KTOT / 32, HIDDEN / 64), 256, 0, stream>>>(W, V, wr_t);

    for (int b0 = 0; b0 < BSZ; b0 += G) {
        const int gc = (BSZ - b0 < G) ? (BSZ - b0) : G;
        aggregate_kernel<<<dim3(NNODE, gc), 256, 0, stream>>>(x_f32, adj, sup_bf, b0);
        rgcn_gemm_kernel<<<dim3(16, gc), 256, 0, stream>>>(x_bf, sup_bf, wr_t, out, b0);
    }
}
